// scPert_Model_36996848288000
// MI455X (gfx1250) — matrix operand order repaired and checked by reading, not compiled, not run on hardware
//
#include <hip/hip_runtime.h>
#include <hip/hip_bf16.h>
#include <math.h>

// ---------------- model dims ----------------
static constexpr int kNG = 5120;
static constexpr int kE  = 512;
static constexpr int kH  = 256;
static constexpr int kNH = 8;
static constexpr int kDK = 32;
static constexpr int kCH = 256;
static constexpr int kNL = 8;
static constexpr int kB  = 4;
static constexpr int kP  = 2;
static constexpr int kFF = 1024;
static constexpr int kNC = kNG / kCH;     // 20 chunks
static constexpr int kTOK = kB * kNG;     // 20480 rows

typedef __attribute__((ext_vector_type(16))) __bf16 v16bf;
typedef __attribute__((ext_vector_type(8)))  float  v8f;
typedef __attribute__((ext_vector_type(4)))  unsigned su4;
typedef __attribute__((ext_vector_type(8)))  unsigned su8;
typedef unsigned short u16;

__device__ __forceinline__ u16 f2bf(float f) {
  unsigned int u = __builtin_bit_cast(unsigned int, f);
  u += 0x7FFFu + ((u >> 16) & 1u);          // round-to-nearest-even
  return (u16)(u >> 16);
}
__device__ __forceinline__ __bf16 bfbits(u16 s) {
  return __builtin_bit_cast(__bf16, s);
}

// low 32b of a generic pointer into LDS = workgroup-relative LDS byte offset
__device__ __forceinline__ unsigned ldsoff(const void* p) {
  return (unsigned)(unsigned long long)p;
}
// CDNA5 async global->LDS copy (16B per lane), tracked by ASYNCcnt.
__device__ __forceinline__ void async_b128(unsigned lds, const u16* g) {
  asm volatile("global_load_async_to_lds_b128 %0, %1, off"
               :: "v"(lds), "v"(g) : "memory");
}
__device__ __forceinline__ void wait_async() {
  asm volatile("s_wait_asynccnt 0" ::: "memory");
}

// ---------------------------------------------------------------------
// Tensor Data Mover: DMA a 2-D tile of bf16 (2-byte) elements from
// global memory into LDS.  D# descriptor per CDNA5 ISA ch.8:
//   group0: count=1 | lds_addr | global_addr | type=2
//   group1: data_size=2B, tensor_dim0/1 (OOB extent), tile_dim0/1,
//           tensor_dim0_stride (elements)
// Issue from ONE wave only; completion via TENSORcnt.
// ---------------------------------------------------------------------
__device__ __forceinline__ void tdm_load_2d(unsigned lds, const void* gaddr,
                                            unsigned tile_x, unsigned tile_y,
                                            unsigned ext_x, unsigned ext_y,
                                            unsigned stride_elems)
{
  unsigned long long ga = (unsigned long long)gaddr;
  su4 g0;
  g0[0] = 1u;                                        // count=1, user mode
  g0[1] = lds;                                       // lds_addr (bytes)
  g0[2] = (unsigned)ga;                              // global_addr[31:0]
  g0[3] = ((unsigned)(ga >> 32) & 0x01FFFFFFu) | 0x80000000u;  // addr[56:32] | type=2
  su8 g1;
  g1[0] = 0x00010000u;                               // data_size=1 (2 bytes)
  g1[1] = (ext_x & 0xFFFFu) << 16;                   // tensor_dim0 lo16
  g1[2] = (ext_x >> 16) | ((ext_y & 0xFFFFu) << 16); // tensor_dim0 hi / dim1 lo
  g1[3] = (ext_y >> 16) | (tile_x << 16);            // tensor_dim1 hi / tile_dim0
  g1[4] = tile_y & 0xFFFFu;                          // tile_dim1 (tile_dim2=0)
  g1[5] = stride_elems;                              // tensor_dim0_stride lo32
  g1[6] = 0u;
  g1[7] = 0u;
  su4 gz = {0u, 0u, 0u, 0u};                         // dims 2+ unused (zero)
  asm volatile("tensor_load_to_lds %0, %1, %2, %3"
               :: "s"(g0), "s"(g1), "s"(gz), "s"(gz) : "memory");
}

// =====================================================================
// GEMM: C[M,N] = act( alpha * A@B + bias ),  A bf16 [M,K],
// B bf16 row-major [K,N] (transB=0) or [N,K] (transB=1).
// K%32==0, N%128==0, M guarded (TDM OOB rows read as zero).
// 256 thr (8 waves), tile 128x128.  LDS operand-major: As[m][k], Bs[n][k].
// A tile is moved by the Tensor Data Mover (wave 0); transB B tile by
// async global->LDS; row-major B tile transposed through VGPRs.
// =====================================================================
__global__ __launch_bounds__(256)
void gemm_bf16_wmma(const u16* __restrict__ A, const u16* __restrict__ Bm,
                    const float* __restrict__ bias, void* __restrict__ Cv,
                    int M, int N, int K, float alpha, int transB, int act, int obf)
{
  __shared__ __align__(16) u16 As[128 * 32];
  __shared__ __align__(16) u16 Bs[128 * 32];

  const int tid  = threadIdx.x;
  const int lane = tid & 31;
  const int wid  = tid >> 5;
  const int wm   = wid & 1;
  const int wn   = wid >> 1;
  const int blockN = blockIdx.x * 128;
  const int blockM = blockIdx.y * 128;
  const int hl  = lane >> 4;
  const int l16 = lane & 15;
  const unsigned ldsA = ldsoff(As);

  v8f acc[4][2];
  #pragma unroll
  for (int a = 0; a < 4; a++)
    #pragma unroll
    for (int b = 0; b < 2; b++)
      #pragma unroll
      for (int e = 0; e < 8; e++) acc[a][b][e] = 0.f;

  for (int k0 = 0; k0 < K; k0 += 32) {
    __syncthreads();
    // ---- A tile 128x32 via Tensor Data Mover (wave 0 issues) ----
    if (wid == 0) {
      tdm_load_2d(ldsA, A + (size_t)blockM * K + k0,
                  /*tile*/32, 128,
                  /*extent*/(unsigned)(K - k0), (unsigned)(M - blockM),
                  /*stride*/(unsigned)K);
    }
    // ---- B tile as Bs[n][k] ----
    if (transB) {
      int c  = tid >> 1;
      int kb = (tid & 1) * 16;
      const u16* g = Bm + (size_t)(blockN + c) * K + k0 + kb;
      unsigned l = ldsoff(&Bs[c * 32 + kb]);
      async_b128(l, g);
      async_b128(l + 16, g + 8);
    } else {
      int kk = tid >> 3;
      int cb = (tid & 7) * 16;
      const uint4* gp = (const uint4*)(Bm + (size_t)(k0 + kk) * N + blockN + cb);
      uint4 d0 = gp[0], d1 = gp[1];
      unsigned v32[8] = {d0.x, d0.y, d0.z, d0.w, d1.x, d1.y, d1.z, d1.w};
      #pragma unroll
      for (int i = 0; i < 8; i++) {
        Bs[(cb + 2*i + 0) * 32 + kk] = (u16)(v32[i] & 0xffffu);
        Bs[(cb + 2*i + 1) * 32 + kk] = (u16)(v32[i] >> 16);
      }
      if (k0 + 32 < K) __builtin_prefetch(Bm + (size_t)(k0 + 32 + kk) * N + blockN + cb, 0, 1);
    }
    wait_async();
    if (wid == 0) __builtin_amdgcn_s_wait_tensorcnt(0);
    __syncthreads();
    // ---- WMMA ----
    v16bf bfrag[2];
    #pragma unroll
    for (int nt = 0; nt < 2; nt++) {
      const u16* bp = &Bs[(wn * 32 + nt * 16 + l16) * 32 + hl * 16];
      #pragma unroll
      for (int e = 0; e < 16; e++) bfrag[nt][e] = bfbits(bp[e]);
    }
    #pragma unroll
    for (int mt = 0; mt < 4; mt++) {
      const u16* ap = &As[(wm * 64 + mt * 16 + l16) * 32 + hl * 16];
      v16bf afrag;
      #pragma unroll
      for (int e = 0; e < 8; e++) { afrag[e] = bfbits(ap[e]); afrag[8 + e] = bfbits(ap[8 + e]); }
      #pragma unroll
      for (int nt = 0; nt < 2; nt++) {
        acc[mt][nt] = __builtin_amdgcn_wmma_f32_16x16x32_bf16(
            false, afrag, false, bfrag[nt], (short)0, acc[mt][nt], false, false);
      }
    }
  }
  // ---- epilogue ----
  #pragma unroll
  for (int mt = 0; mt < 4; mt++) {
    #pragma unroll
    for (int nt = 0; nt < 2; nt++) {
      #pragma unroll
      for (int r = 0; r < 8; r++) {
        int row = blockM + wm * 64 + mt * 16 + r + hl * 8;
        int col = blockN + wn * 32 + nt * 16 + l16;
        if (row < M) {
          float v = acc[mt][nt][r] * alpha;
          if (bias) v += bias[col];
          if (act == 1)      v = v > 0.f ? v : 0.f;
          else if (act == 2) v = 0.5f * v * (1.f + erff(v * 0.70710678118654752f));
          if (obf) ((u16*)Cv)[(size_t)row * N + col] = f2bf(v);
          else     ((float*)Cv)[(size_t)row * N + col] = v;
        }
      }
    }
  }
}

// =====================================================================
// LayerNorm; optional residual add, relu, f32 and/or bf16 outputs.
// =====================================================================
__global__ __launch_bounds__(256)
void ln_kernel(const float* __restrict__ X, const float* __restrict__ Res,
               const float* __restrict__ g, const float* __restrict__ b,
               float* __restrict__ Y, u16* __restrict__ Ybf, int W, int relu)
{
  __shared__ float rs[8], rss[8], stat[2];
  const int row = blockIdx.x;
  const int nv  = W >> 8;
  float vals[4];
  float s = 0.f, ss = 0.f;
  for (int i = 0; i < nv; i++) {
    int c = threadIdx.x + (i << 8);
    float v = X[(size_t)row * W + c];
    if (Res) v += Res[(size_t)row * W + c];
    vals[i] = v; s += v; ss += v * v;
  }
  for (int off = 16; off > 0; off >>= 1) {
    s  += __shfl_xor(s, off, 32);
    ss += __shfl_xor(ss, off, 32);
  }
  if ((threadIdx.x & 31) == 0) { rs[threadIdx.x >> 5] = s; rss[threadIdx.x >> 5] = ss; }
  __syncthreads();
  if (threadIdx.x == 0) {
    float a = 0.f, c2 = 0.f;
    for (int i = 0; i < 8; i++) { a += rs[i]; c2 += rss[i]; }
    stat[0] = a; stat[1] = c2;
  }
  __syncthreads();
  float mu   = stat[0] / (float)W;
  float var  = stat[1] / (float)W - mu * mu;
  float rstd = rsqrtf(var + 1e-5f);
  for (int i = 0; i < nv; i++) {
    int c = threadIdx.x + (i << 8);
    float v = (vals[i] - mu) * rstd * g[c] + b[c];
    if (relu) v = v > 0.f ? v : 0.f;
    if (Y)   Y[(size_t)row * W + c] = v;
    if (Ybf) Ybf[(size_t)row * W + c] = f2bf(v);
  }
}

// =====================================================================
// In-place row softmax (f32) + bf16 copy of the probabilities.
// =====================================================================
__global__ __launch_bounds__(256)
void softmax_rows(float* __restrict__ S, u16* __restrict__ Sbf, int W)
{
  __shared__ float rbuf[8];
  __shared__ float bval;
  const int row = blockIdx.x;
  float* p = S + (size_t)row * W;
  float m = -3.4e38f;
  for (int c = threadIdx.x; c < W; c += 256) m = fmaxf(m, p[c]);
  for (int off = 16; off > 0; off >>= 1) m = fmaxf(m, __shfl_xor(m, off, 32));
  if ((threadIdx.x & 31) == 0) rbuf[threadIdx.x >> 5] = m;
  __syncthreads();
  if (threadIdx.x == 0) { float a = rbuf[0]; for (int i = 1; i < 8; i++) a = fmaxf(a, rbuf[i]); bval = a; }
  __syncthreads();
  m = bval;
  float s = 0.f;
  for (int c = threadIdx.x; c < W; c += 256) { float e = __expf(p[c] - m); p[c] = e; s += e; }
  for (int off = 16; off > 0; off >>= 1) s += __shfl_xor(s, off, 32);
  if ((threadIdx.x & 31) == 0) rbuf[threadIdx.x >> 5] = s;
  __syncthreads();
  if (threadIdx.x == 0) { float a = 0.f; for (int i = 0; i < 8; i++) a += rbuf[i]; bval = a; }
  __syncthreads();
  float rinv = 1.f / bval;
  for (int c = threadIdx.x; c < W; c += 256) {
    float v = p[c] * rinv;
    p[c] = v;
    Sbf[(size_t)row * W + c] = f2bf(v);
  }
}

// =====================================================================
// Fused block-local attention (bf16 in / bf16 out).
// grid = B*NH*NC, block = 128 (4 waves).
// K tile staged by TDM; V tile transposed through VGPRs.
// =====================================================================
__global__ __launch_bounds__(128)
void chunk_attn(const u16* __restrict__ Q, const u16* __restrict__ Km,
                const u16* __restrict__ V, u16* __restrict__ O)
{
  __shared__ __align__(16) u16 ks[kCH * kDK];      // [kv_row][d]  16 KB
  __shared__ __align__(16) u16 vs[kDK * kCH];      // [d][kv_row]  16 KB (transposed)
  __shared__ __align__(16) u16 ps[4 * 16 * kCH];   // per-wave P    32 KB

  const int bid = blockIdx.x;
  const int c   = bid % kNC;
  const int h   = (bid / kNC) % kNH;
  const int b   = bid / (kNC * kNH);
  const size_t rowbase = (size_t)b * kNG + (size_t)c * kCH;

  const int tid  = threadIdx.x;
  const int lane = tid & 31;
  const int w    = tid >> 5;
  const int hl   = lane >> 4;
  const int l16  = lane & 15;

  // K tile (256 rows x 32 bf16, row stride kH) via Tensor Data Mover
  if (w == 0) {
    tdm_load_2d(ldsoff(ks), Km + rowbase * kH + h * kDK,
                /*tile*/kDK, kCH, /*extent*/kDK, kCH, /*stride*/kH);
  }
  // V tile staged transposed: vs[d][kv_row]
  for (int i = tid; i < kCH * 4; i += 128) {
    int r = i >> 2, part = i & 3;
    const uint4* vg = (const uint4*)(V + (rowbase + r) * kH + h * kDK);
    uint4 d = vg[part];
    unsigned v32[4] = {d.x, d.y, d.z, d.w};
    #pragma unroll
    for (int j = 0; j < 4; j++) {
      vs[(part * 8 + 2*j + 0) * kCH + r] = (u16)(v32[j] & 0xffffu);
      vs[(part * 8 + 2*j + 1) * kCH + r] = (u16)(v32[j] >> 16);
    }
  }
  if (w == 0) __builtin_amdgcn_s_wait_tensorcnt(0);
  __syncthreads();

  u16* pw = ps + w * 16 * kCH;
  const float scale = 0.17677669529663687f;   // 1/sqrt(DK)

  for (int t = w; t < 16; t += 4) {
    const int m0 = t * 16;
    v16bf aq;
    {
      const u16* qp = Q + (rowbase + m0 + l16) * kH + h * kDK + hl * 16;
      #pragma unroll
      for (int e = 0; e < 8; e++) { aq[e] = bfbits(qp[e]); aq[8 + e] = bfbits(qp[8 + e]); }
    }
    // S = q @ k^T
    v8f z;
    #pragma unroll
    for (int e = 0; e < 8; e++) z[e] = 0.f;
    v8f sacc[16];
    #pragma unroll
    for (int j = 0; j < 16; j++) {
      const u16* bp = &ks[(j * 16 + l16) * kDK + hl * 16];
      v16bf bk;
      #pragma unroll
      for (int e = 0; e < 16; e++) bk[e] = bfbits(bp[e]);
      sacc[j] = __builtin_amdgcn_wmma_f32_16x16x32_bf16(false, aq, false, bk, (short)0, z, false, false);
    }
    // softmax across 256 cols (rows live in 16-lane halves)
    float rmax[8], rsum[8];
    #pragma unroll
    for (int r = 0; r < 8; r++) {
      float mx = -3.4e38f;
      #pragma unroll
      for (int j = 0; j < 16; j++) mx = fmaxf(mx, sacc[j][r]);
      for (int off = 1; off < 16; off <<= 1) mx = fmaxf(mx, __shfl_xor(mx, off, 32));
      rmax[r] = mx;
    }
    #pragma unroll
    for (int r = 0; r < 8; r++) {
      float sm = 0.f;
      #pragma unroll
      for (int j = 0; j < 16; j++) {
        float e = __expf((sacc[j][r] - rmax[r]) * scale);
        sacc[j][r] = e; sm += e;
      }
      for (int off = 1; off < 16; off <<= 1) sm += __shfl_xor(sm, off, 32);
      rsum[r] = sm;
    }
    #pragma unroll
    for (int r = 0; r < 8; r++) {
      float rinv = 1.f / rsum[r];
      int prow = r + hl * 8;
      #pragma unroll
      for (int j = 0; j < 16; j++)
        pw[prow * kCH + j * 16 + l16] = f2bf(sacc[j][r] * rinv);
    }
    // O = P @ V
    v8f oacc[2];
    #pragma unroll
    for (int nt = 0; nt < 2; nt++)
      #pragma unroll
      for (int e = 0; e < 8; e++) oacc[nt][e] = 0.f;
    for (int s8 = 0; s8 < 8; s8++) {
      const u16* app = pw + l16 * kCH + s8 * 32 + hl * 16;
      v16bf ap;
      #pragma unroll
      for (int e = 0; e < 8; e++) { ap[e] = bfbits(app[e]); ap[8 + e] = bfbits(app[8 + e]); }
      #pragma unroll
      for (int nt = 0; nt < 2; nt++) {
        const u16* bp = vs + (nt * 16 + l16) * kCH + s8 * 32 + hl * 16;
        v16bf bv;
        #pragma unroll
        for (int e = 0; e < 16; e++) bv[e] = bfbits(bp[e]);
        oacc[nt] = __builtin_amdgcn_wmma_f32_16x16x32_bf16(false, ap, false, bv, (short)0, oacc[nt], false, false);
      }
    }
    #pragma unroll
    for (int nt = 0; nt < 2; nt++)
      #pragma unroll
      for (int r = 0; r < 8; r++) {
        int grow = m0 + r + hl * 8;
        int d = nt * 16 + l16;
        O[(rowbase + grow) * kH + h * kDK + d] = f2bf(oacc[nt][r]);
      }
  }
}

// ---------------- small elementwise kernels ----------------
__global__ __launch_bounds__(256)
void cvt_f32_bf16(const float* __restrict__ x, u16* __restrict__ y, int n)
{
  int i = blockIdx.x * 256 + threadIdx.x;
  if (i < n) y[i] = f2bf(x[i]);
}

__global__ __launch_bounds__(256)
void pert_finalize(const float* __restrict__ p6, const float* __restrict__ ctrl,
                   const float* __restrict__ escale, float* __restrict__ pm)
{
  __shared__ float rb[8]; __shared__ float tot;
  int b = blockIdx.x, c = threadIdx.x;
  float fe = 0.5f * (p6[(b * kP + 0) * kH + c] + p6[(b * kP + 1) * kH + c]);
  float s = fe;
  for (int off = 16; off > 0; off >>= 1) s += __shfl_xor(s, off, 32);
  if ((c & 31) == 0) rb[c >> 5] = s;
  __syncthreads();
  if (c == 0) { float a = 0.f; for (int i = 0; i < 8; i++) a += rb[i]; tot = a; }
  __syncthreads();
  float v = (tot == 0.0f) ? ctrl[c] : fe;
  pm[b * kH + c] = v * escale[0];
}

__global__ __launch_bounds__(256)
void add_bcast(const float* __restrict__ gh, const float* __restrict__ pm,
               float* __restrict__ out)
{
  size_t idx = (size_t)blockIdx.x * 256 + threadIdx.x;
  int cc = (int)(idx & (kH - 1));
  size_t row = idx >> 8;
  int b = (int)(row / kNG);
  int g = (int)(row % kNG);
  out[idx] = gh[(size_t)g * kH + cc] + pm[b * kH + cc];
}

__global__ __launch_bounds__(256)
void vadd(const float* __restrict__ a, const float* __restrict__ b, float* __restrict__ c)
{
  size_t i = (size_t)blockIdx.x * 256 + threadIdx.x;
  c[i] = a[i] + b[i];
}

// src += 0.5*s  (f32), plus bf16 mirror of updated src
__global__ __launch_bounds__(256)
void axpy_half(float* __restrict__ dst, const float* __restrict__ s, u16* __restrict__ dbf)
{
  size_t i = (size_t)blockIdx.x * 256 + threadIdx.x;
  float v = dst[i] + 0.5f * s[i];
  dst[i] = v;
  dbf[i] = f2bf(v);
}

__global__ __launch_bounds__(256)
void out_head(const float* __restrict__ Hm, const float* __restrict__ w,
              const float* __restrict__ bb, const float* __restrict__ x,
              float* __restrict__ out)
{
  int row  = blockIdx.x * 8 + (threadIdx.x >> 5);
  int lane = threadIdx.x & 31;
  float s = 0.f;
  for (int i = lane; i < kH; i += 32) s += Hm[(size_t)row * kH + i] * w[i];
  for (int off = 16; off > 0; off >>= 1) s += __shfl_xor(s, off, 32);
  if (lane == 0) out[row] = s + bb[0] + x[row];
}

// =====================================================================
// Host orchestration
// =====================================================================
extern "C" void kernel_launch(void* const* d_in, const int* in_sizes, int n_in,
                              void* d_out, int out_size, void* d_ws, size_t ws_size,
                              hipStream_t stream)
{
  (void)in_sizes; (void)n_in; (void)out_size; (void)ws_size;
  int i = 0;
  const float* in_x     = (const float*)d_in[i++];
  const float* in_gene  = (const float*)d_in[i++];
  const float* in_pert  = (const float*)d_in[i++];
  const float* gd_lin_w = (const float*)d_in[i++];
  const float* gd_lin_b = (const float*)d_in[i++];
  const float* gd_ln_g  = (const float*)d_in[i++];
  const float* gd_ln_b  = (const float*)d_in[i++];
  const float* gi_lin_w = (const float*)d_in[i++];
  const float* gi_lin_b = (const float*)d_in[i++];
  const float* gi_ln_g  = (const float*)d_in[i++];
  const float* gi_ln_b  = (const float*)d_in[i++];
  const float* pd_l1_w  = (const float*)d_in[i++];
  const float* pd_l1_b  = (const float*)d_in[i++];
  const float* pd_ln_g  = (const float*)d_in[i++];
  const float* pd_ln_b  = (const float*)d_in[i++];
  const float* pd_l2_w  = (const float*)d_in[i++];
  const float* pd_l2_b  = (const float*)d_in[i++];
  const float* pf_l1_w  = (const float*)d_in[i++];
  const float* pf_l1_b  = (const float*)d_in[i++];
  const float* pf_ln1_g = (const float*)d_in[i++];
  const float* pf_ln1_b = (const float*)d_in[i++];
  const float* pf_l2_w  = (const float*)d_in[i++];
  const float* pf_l2_b  = (const float*)d_in[i++];
  const float* pf_ln2_g = (const float*)d_in[i++];
  const float* pf_ln2_b = (const float*)d_in[i++];
  const float* emb_scale= (const float*)d_in[i++];
  const float* ctrl_emb = (const float*)d_in[i++];
  const float* lnin_g   = (const float*)d_in[i++];
  const float* lnin_b   = (const float*)d_in[i++];
  const float* lnout_g  = (const float*)d_in[i++];
  const float* lnout_b  = (const float*)d_in[i++];
  const float* f_l1_w   = (const float*)d_in[i++];
  const float* f_l1_b   = (const float*)d_in[i++];
  const float* f_n1_g   = (const float*)d_in[i++];
  const float* f_n1_b   = (const float*)d_in[i++];
  const float* f_l2_w   = (const float*)d_in[i++];
  const float* f_l2_b   = (const float*)d_in[i++];
  const float* f_n2_g   = (const float*)d_in[i++];
  const float* f_n2_b   = (const float*)d_in[i++];
  const float* f_l3_w   = (const float*)d_in[i++];
  const float* f_l3_b   = (const float*)d_in[i++];
  const float* out_w    = (const float*)d_in[i++];
  const float* out_b    = (const float*)d_in[i++];
  const float* Lq_w  = (const float*)d_in[i++];
  const float* Lq_b  = (const float*)d_in[i++];
  const float* Lk_w  = (const float*)d_in[i++];
  const float* Lk_b  = (const float*)d_in[i++];
  const float* Lv_w  = (const float*)d_in[i++];
  const float* Lv_b  = (const float*)d_in[i++];
  const float* Lo_w  = (const float*)d_in[i++];
  const float* Lo_b  = (const float*)d_in[i++];
  const float* Lf1_w = (const float*)d_in[i++];
  const float* Lf1_b = (const float*)d_in[i++];
  const float* Lf2_w = (const float*)d_in[i++];
  const float* Lf2_b = (const float*)d_in[i++];
  const float* Ln1_g = (const float*)d_in[i++];
  const float* Ln1_b = (const float*)d_in[i++];
  const float* Ln2_g = (const float*)d_in[i++];
  const float* Ln2_b = (const float*)d_in[i++];

  // ---- workspace bump allocation ----
  char* base = (char*)d_ws;
  size_t off = 0;
  auto allocF = [&](size_t n) { float* p = (float*)(base + off); off = (off + n * 4 + 255) & ~(size_t)255; return p; };
  auto allocH = [&](size_t n) { u16*   p = (u16*)(base + off);   off = (off + n * 2 + 255) & ~(size_t)255; return p; };

  float* p_gh   = allocF(1310720);
  float* p_gh2  = allocF(1310720);
  float* p_tmp  = allocF(1310720);
  float* p_t    = allocF(1310720);
  float* p_sc   = allocF(26214400);   // scores / srcpre / final h1
  float* p_p1   = allocF(4096);
  float* p_p4   = allocF(2048);
  float* p_p5   = allocF(2048);
  float* p_pm   = allocF(1024);
  float* p_src  = allocF(5242880);
  float* p_s1   = allocF(5242880);
  float* p_ao   = allocF(5242880);    // attn out-proj / ff2 out / final h2
  float* p_v    = allocF(5242880);    // block out / final h3
  u16* w_gene = allocH(2621440);
  u16* w_pert = allocH(5120);
  u16* w_gd   = allocH(131072);
  u16* w_gi   = allocH(65536);
  u16* w_pd1  = allocH(327680);
  u16* w_pd2  = allocH(131072);
  u16* w_pf1  = allocH(131072);
  u16* w_pf2  = allocH(131072);
  u16* w_fl1  = allocH(262144);
  u16* w_fl2  = allocH(524288);
  u16* w_fl3  = allocH(131072);
  u16* w_q    = allocH(524288);
  u16* w_k    = allocH(524288);
  u16* w_v    = allocH(524288);
  u16* w_o    = allocH(524288);
  u16* w_f1   = allocH(2097152);
  u16* w_f2   = allocH(2097152);
  u16* gh_bf   = allocH(1310720);
  u16* probs_bf= allocH(26214400);    // probs / final h2b
  u16* ctx_bf  = allocH(1310720);
  u16* p2bf    = allocH(4096);
  u16* p3bf    = allocH(2048);
  u16* src_bf  = allocH(5242880);
  u16* s1_bf   = allocH(5242880);
  u16* q_bf    = allocH(5242880);
  u16* k_bf    = allocH(5242880);
  u16* v_bf    = allocH(5242880);
  u16* at_bf   = allocH(5242880);
  u16* ff_bf   = allocH(20971520);    // ff1 / final h1b

  auto cvt = [&](const float* s, u16* d, int n) {
    cvt_f32_bf16<<<dim3((n + 255) / 256), dim3(256), 0, stream>>>(s, d, n);
  };
  auto gemm = [&](const u16* A, const u16* Bm, const float* bias, void* C,
                  int M, int N, int K, float alpha, int transB, int act, int obf) {
    dim3 g(N / 128, (M + 127) / 128);
    gemm_bf16_wmma<<<g, dim3(256), 0, stream>>>(A, Bm, bias, C, M, N, K, alpha, transB, act, obf);
  };
  auto ln = [&](const float* X, const float* Res, const float* g, const float* b,
                float* Y, u16* Ybf, int W, int rows, int relu) {
    ln_kernel<<<dim3(rows), dim3(256), 0, stream>>>(X, Res, g, b, Y, Ybf, W, relu);
  };

  // ---- one-time (per launch) bf16 conversion of embeddings + weights ----
  cvt(in_gene, w_gene, 2621440);
  cvt(in_pert, w_pert, 5120);
  cvt(gd_lin_w, w_gd, 131072);
  cvt(gi_lin_w, w_gi, 65536);
  cvt(pd_l1_w, w_pd1, 327680);
  cvt(pd_l2_w, w_pd2, 131072);
  cvt(pf_l1_w, w_pf1, 131072);
  cvt(pf_l2_w, w_pf2, 131072);
  cvt(f_l1_w, w_fl1, 262144);
  cvt(f_l2_w, w_fl2, 524288);
  cvt(f_l3_w, w_fl3, 131072);
  cvt(Lq_w, w_q, 524288);
  cvt(Lk_w, w_k, 524288);
  cvt(Lv_w, w_v, 524288);
  cvt(Lo_w, w_o, 524288);
  cvt(Lf1_w, w_f1, 2097152);
  cvt(Lf2_w, w_f2, 2097152);

  // ================= gene branch =================
  gemm(w_gene, w_gd, gd_lin_b, p_tmp, kNG, kH, kE, 1.f, 0, 0, 0);
  ln(p_tmp, nullptr, gd_ln_g, gd_ln_b, p_gh, gh_bf, kH, kNG, 1);
  gemm(gh_bf, gh_bf, nullptr, p_sc, kNG, kNG, kH, 0.0625f, 1, 0, 0);
  softmax_rows<<<dim3(kNG), dim3(256), 0, stream>>>(p_sc, probs_bf, kNG);
  gemm(probs_bf, gh_bf, nullptr, ctx_bf, kNG, kH, kNG, 1.f, 0, 0, 1);
  gemm(ctx_bf, w_gi, gi_lin_b, p_tmp, kNG, kH, kH, 1.f, 0, 0, 0);
  ln(p_tmp, nullptr, gi_ln_g, gi_ln_b, p_t, nullptr, kH, kNG, 1);
  vadd<<<dim3(1310720 / 256), dim3(256), 0, stream>>>(p_gh, p_t, p_gh2);

  // ================= pert branch =================
  gemm(w_pert, w_pd1, pd_l1_b, p_p1, kB * kP, 2 * kH, 640, 1.f, 0, 0, 0);
  ln(p_p1, nullptr, pd_ln_g, pd_ln_b, nullptr, p2bf, 2 * kH, kB * kP, 1);
  gemm(p2bf, w_pd2, pd_l2_b, p3bf, kB * kP, kH, 2 * kH, 1.f, 0, 0, 1);
  gemm(p3bf, w_pf1, pf_l1_b, p_p1, kB * kP, 2 * kH, kH, 1.f, 0, 0, 0);
  ln(p_p1, nullptr, pf_ln1_g, pf_ln1_b, nullptr, p2bf, 2 * kH, kB * kP, 1);
  gemm(p2bf, w_pf2, pf_l2_b, p_p4, kB * kP, kH, 2 * kH, 1.f, 0, 0, 0);
  ln(p_p4, nullptr, pf_ln2_g, pf_ln2_b, p_p5, nullptr, kH, kB * kP, 0);
  pert_finalize<<<dim3(kB), dim3(256), 0, stream>>>(p_p5, ctrl_emb, emb_scale, p_pm);

  add_bcast<<<dim3(kTOK), dim3(256), 0, stream>>>(p_gh2, p_pm, p_sc);
  ln(p_sc, nullptr, lnin_g, lnin_b, p_src, src_bf, kH, kTOK, 0);

  // ================= transformer layers =================
  for (int li = 0; li < kNL; li++) {
    const u16* qw = w_q + (size_t)li * kH * kH;   const float* qb = Lq_b + li * kH;
    const u16* kw = w_k + (size_t)li * kH * kH;   const float* kb = Lk_b + li * kH;
    const u16* vw = w_v + (size_t)li * kH * kH;   const float* vb = Lv_b + li * kH;
    const u16* ow = w_o + (size_t)li * kH * kH;   const float* ob = Lo_b + li * kH;
    const u16* f1w = w_f1 + (size_t)li * kH * kFF; const float* f1b = Lf1_b + li * kFF;
    const u16* f2w = w_f2 + (size_t)li * kFF * kH; const float* f2b = Lf2_b + li * kH;
    const float* n1g = Ln1_g + li * kH; const float* n1b = Ln1_b + li * kH;
    const float* n2g = Ln2_g + li * kH; const float* n2b = Ln2_b + li * kH;

    gemm(src_bf, qw, qb, q_bf, kTOK, kH, kH, 1.f, 0, 0, 1);
    gemm(src_bf, kw, kb, k_bf, kTOK, kH, kH, 1.f, 0, 0, 1);
    gemm(src_bf, vw, vb, v_bf, kTOK, kH, kH, 1.f, 0, 0, 1);
    chunk_attn<<<dim3(kB * kNH * kNC), dim3(128), 0, stream>>>(q_bf, k_bf, v_bf, at_bf);
    gemm(at_bf, ow, ob, p_ao, kTOK, kH, kH, 1.f, 0, 0, 0);
    ln(p_ao, p_src, n1g, n1b, p_s1, s1_bf, kH, kTOK, 0);
    gemm(s1_bf, f1w, f1b, ff_bf, kTOK, kFF, kH, 1.f, 0, 2, 1);    // FF1 + GELU
    gemm(ff_bf, f2w, f2b, p_ao, kTOK, kH, kFF, 1.f, 0, 0, 0);     // FF2
    ln(p_ao, p_s1, n2g, n2b, p_v, nullptr, kH, kTOK, 0);
    axpy_half<<<dim3(kTOK), dim3(256), 0, stream>>>(p_src, p_v, src_bf);
  }

  // ================= final head =================
  ln(p_src, nullptr, lnout_g, lnout_b, nullptr, s1_bf, kH, kTOK, 0);
  gemm(s1_bf, w_fl1, f_l1_b, p_sc, kTOK, 4 * kH, kH, 1.f, 0, 0, 0);
  ln(p_sc, nullptr, f_n1_g, f_n1_b, nullptr, ff_bf, 4 * kH, kTOK, 1);
  gemm(ff_bf, w_fl2, f_l2_b, p_ao, kTOK, 2 * kH, 4 * kH, 1.f, 0, 0, 0);
  ln(p_ao, nullptr, f_n2_g, f_n2_b, nullptr, probs_bf, 2 * kH, kTOK, 1);
  gemm(probs_bf, w_fl3, f_l3_b, p_v, kTOK, kH, 2 * kH, 1.f, 0, 0, 0);
  out_head<<<dim3(kTOK / 8), dim3(256), 0, stream>>>(p_v, out_w, out_b, in_x, (float*)d_out);
}
